// BatchedTorchParametricSolver_81767587381595
// MI455X (gfx1250) — compile-verified, matrix operand order checked by reading, not run on hardware
//
#include <hip/hip_runtime.h>
#include <hip/hip_bf16.h>

// ---------------- problem constants ----------------
#define Hc     256
#define Wc     256
#define Kc     3
#define HWc    (Hc * Wc)          // 65536
#define NTOT   (Kc * HWc)         // 196608 labels per sample
#define LANEW  8
#define NROWS  (NTOT / LANEW)     // 24576
#define NOPSc  HWc                // 65536
#define Bc     8
#define LEN1   262144             // next pow2 >= NTOT for bitonic sort
#define LEN2   65536              // NOPS is already a power of two
#define LOCELEMS 4096             // elements per block for LDS bitonic phases

typedef __attribute__((ext_vector_type(16))) __bf16 v16bf;
typedef __attribute__((ext_vector_type(8)))  float  v8f;

struct Pack32 { uint4 a, b; };    // 32B container for v16bf bit-cast

// ---------------- small helpers ----------------
__device__ __forceinline__ unsigned short f2bf(float f) {
  unsigned int u = __float_as_uint(f);
  unsigned int r = u + 0x7FFFu + ((u >> 16) & 1u);   // round-to-nearest-even
  return (unsigned short)(r >> 16);
}
__device__ __forceinline__ float bf2f(unsigned short u) {
  return __uint_as_float(((unsigned int)u) << 16);
}
__device__ __forceinline__ __bf16 us2bf(unsigned short u) {
  return __builtin_bit_cast(__bf16, u);
}
__device__ __forceinline__ unsigned int f2key(float f) {
  unsigned int u = __float_as_uint(f);
  return (u & 0x80000000u) ? ~u : (u | 0x80000000u);
}

// ---------------- sort kernels ----------------
__global__ void init_keys1_kernel(const float* __restrict__ logits,
                                  unsigned long long* __restrict__ keys) {
  size_t gid = (size_t)blockIdx.x * blockDim.x + threadIdx.x;   // B*LEN1
  int i = (int)(gid % LEN1);
  int b = (int)(gid / LEN1);
  unsigned long long key;
  if (i < NTOT) {
    unsigned int u = f2key(logits[(size_t)b * NTOT + i]);
    key = ((unsigned long long)u << 32) | (unsigned int)i;
  } else {
    key = ~0ull;                                                // pad to tail
  }
  keys[gid] = key;
}

__global__ void init_keys2_kernel(const float* __restrict__ logits,
                                  unsigned long long* __restrict__ keys) {
  size_t gid = (size_t)blockIdx.x * blockDim.x + threadIdx.x;   // B*LEN2
  int i = (int)(gid % LEN2);
  unsigned int u = f2key(logits[gid]);
  keys[gid] = ((unsigned long long)u << 32) | (unsigned int)i;
}

// global compare-swap step (only for j >= LOCELEMS)
__global__ void bitonic_step_kernel(unsigned long long* __restrict__ keys,
                                    int len, unsigned int j, unsigned int k) {
  size_t gid = (size_t)blockIdx.x * blockDim.x + threadIdx.x;   // B*len
  unsigned int i = (unsigned int)(gid % len);
  size_t base = (gid / len) * (size_t)len;
  unsigned int ixj = i ^ j;
  if (ixj > i) {
    unsigned long long a = keys[base + i];
    unsigned long long c = keys[base + ixj];
    bool asc = ((i & k) == 0);
    if ((a > c) == asc) { keys[base + i] = c; keys[base + ixj] = a; }
  }
}

// LDS-resident full bitonic sort of 4096-element chunks (stages k = 2..4096)
__global__ __launch_bounds__(256) void bitonic_sort_local_kernel(
    unsigned long long* __restrict__ keys, int len) {
  __shared__ unsigned long long sh[LOCELEMS];
  size_t base = (size_t)blockIdx.x * LOCELEMS;
  for (int i = threadIdx.x; i < LOCELEMS; i += 256) sh[i] = keys[base + i];
  for (unsigned int k = 2; k <= LOCELEMS; k <<= 1) {
    for (int jl = 31 - __builtin_clz(k >> 1); jl >= 0; --jl) {
      unsigned int j = 1u << jl;
      __syncthreads();
      for (int t = threadIdx.x; t < LOCELEMS / 2; t += 256) {
        unsigned int i = (((unsigned)t >> jl) << (jl + 1)) | ((unsigned)t & (j - 1));
        unsigned int p = i + j;
        unsigned int il = (unsigned int)((base + i) % (unsigned)len);  // batch-local idx
        bool asc = ((il & k) == 0);
        unsigned long long a = sh[i], c = sh[p];
        if ((a > c) == asc) { sh[i] = c; sh[p] = a; }
      }
    }
  }
  __syncthreads();
  for (int i = threadIdx.x; i < LOCELEMS; i += 256) keys[base + i] = sh[i];
}

// LDS-resident merge tail: j = 2048..1 for a fixed k
__global__ __launch_bounds__(256) void bitonic_merge_local_kernel(
    unsigned long long* __restrict__ keys, int len, unsigned int k) {
  __shared__ unsigned long long sh[LOCELEMS];
  size_t base = (size_t)blockIdx.x * LOCELEMS;
  for (int i = threadIdx.x; i < LOCELEMS; i += 256) sh[i] = keys[base + i];
  for (int jl = 11; jl >= 0; --jl) {
    unsigned int j = 1u << jl;
    __syncthreads();
    for (int t = threadIdx.x; t < LOCELEMS / 2; t += 256) {
      unsigned int i = (((unsigned)t >> jl) << (jl + 1)) | ((unsigned)t & (j - 1));
      unsigned int p = i + j;
      unsigned int il = (unsigned int)((base + i) % (unsigned)len);
      bool asc = ((il & k) == 0);
      unsigned long long a = sh[i], c = sh[p];
      if ((a > c) == asc) { sh[i] = c; sh[p] = a; }
    }
  }
  __syncthreads();
  for (int i = threadIdx.x; i < LOCELEMS; i += 256) keys[base + i] = sh[i];
}

__global__ void extract_lo_kernel(const unsigned long long* __restrict__ keys,
                                  unsigned int* __restrict__ out,
                                  int len_keys, int len_out) {
  size_t gid = (size_t)blockIdx.x * blockDim.x + threadIdx.x;   // B*len_out
  int r = (int)(gid % len_out);
  int b = (int)(gid / len_out);
  out[gid] = (unsigned int)keys[(size_t)b * len_keys + r];
}

// ---------------- conv1 (grouped 1->8) fused with permutation scatter ----------------
__global__ void conv1_scatter_kernel(const unsigned int* __restrict__ addr,
                                     const float* __restrict__ w1,
                                     const float* __restrict__ b1,
                                     unsigned short* __restrict__ mem) {
  size_t gid = (size_t)blockIdx.x * blockDim.x + threadIdx.x;   // B*K*HW
  int t  = (int)(gid % HWc);
  int bk = (int)(gid / HWc);
  int k  = bk % Kc;
  int b  = bk / Kc;
  int y = t >> 8, x = t & 255;
  const unsigned int* ab = addr + (size_t)b * NTOT + (size_t)k * HWc;
  float acc[8];
#pragma unroll
  for (int cc = 0; cc < 8; ++cc) acc[cc] = b1[k * 8 + cc];
#pragma unroll
  for (int dy = -1; dy <= 1; ++dy) {
#pragma unroll
    for (int dx = -1; dx <= 1; ++dx) {
      int yy = y + dy, xx = x + dx;
      if (yy < 0 || yy >= Hc || xx < 0 || xx >= Wc) continue;
      float xv = (float)ab[yy * Wc + xx];
      int tap = (dy + 1) * 3 + (dx + 1);
#pragma unroll
      for (int cc = 0; cc < 8; ++cc)
        acc[cc] = fmaf(xv, w1[(k * 8 + cc) * 9 + tap], acc[cc]);
    }
  }
  unsigned int dst = ab[t];
  unsigned short* mb = mem + (size_t)b * 8 * NTOT;
#pragma unroll
  for (int cc = 0; cc < 8; ++cc) {
    float v = acc[cc] > 0.f ? acc[cc] : 0.f;
    mb[(size_t)cc * NTOT + dst] = f2bf(v);
  }
}

// ---------------- conv2 (8->16, 3x3 on [NROWS,8]) as LDS-staged implicit GEMM ----------------
// Phase 1: async-DMA the halo tile to LDS.  Phase 2: build im2col bmat + packed weights
// in LDS with uniform branch-free arithmetic.  Phase 3: fragments are contiguous 16B
// ds_load_b128 runs -> v_wmma_f32_16x16x32_bf16, EXEC untouched.
__global__ __launch_bounds__(256) void conv2_wmma_kernel(
    const unsigned short* __restrict__ mem,
    const float* __restrict__ w2, const float* __restrict__ b2,
    unsigned short* __restrict__ mem2) {
  __shared__ __align__(16) unsigned short tile[8][18][8];     // [ic][r-1..r+16][col]
  __shared__ __align__(16) unsigned short wA[3][16][32];      // [kstep][m][K%32]
  __shared__ __align__(16) unsigned short bmat[8][3][16][32]; // [wave][kstep][n][K%32]

  int blk = blockIdx.x;                        // B * (NROWS/16)
  int b  = blk / (NROWS / 16);
  int r0 = (blk % (NROWS / 16)) * 16;
  const unsigned short* memb = mem + (size_t)b * 8 * NTOT;

  // ---- phase 1: stage input tile (8 ic x 18 rows, 16B per row) via async DMA ----
  for (int idx = threadIdx.x; idx < 8 * 18; idx += 256) {
    int ic = idx / 18, lr = idx % 18;
    int rr = r0 - 1 + lr;
    unsigned short* dst = &tile[ic][lr][0];
    if (rr >= 0 && rr < NROWS) {
      const unsigned short* src = memb + (size_t)ic * NTOT + (size_t)rr * 8;
      unsigned int lds_off = (unsigned int)(uintptr_t)dst;
      unsigned long long ga = (unsigned long long)(uintptr_t)src;
      asm volatile("global_load_async_to_lds_b128 %0, %1, off"
                   :: "v"(lds_off), "v"(ga) : "memory");
    } else {
      *(uint4*)dst = make_uint4(0u, 0u, 0u, 0u);              // zero halo
    }
  }
  // pack weights while DMA is in flight: wA[s][m][kk], q = 32s+kk = ic*12 + tap
  for (int idx = threadIdx.x; idx < 3 * 16 * 32; idx += 256) {
    int kk = idx & 31, m = (idx >> 5) & 15, s = idx >> 9;
    int q = s * 32 + kk;
    int ic = q / 12, tap = q - ic * 12;
    wA[s][m][kk] = (tap < 9) ? f2bf(w2[(m * 8 + ic) * 9 + tap]) : (unsigned short)0;
  }
  asm volatile("s_wait_asynccnt 0" ::: "memory");
  __syncthreads();

  // ---- phase 2: build im2col matrix in LDS (branch-free) ----
  for (int idx = threadIdx.x; idx < 8 * 3 * 16 * 32; idx += 256) {
    int kk = idx & 31;
    int n  = (idx >> 5) & 15;
    int rem = idx >> 9;                        // w*3 + s
    int s  = rem % 3;
    int w  = rem / 3;
    int q  = s * 32 + kk;
    int ic = q / 12, tap = q - ic * 12;
    int ok = tap < 9;
    int tp = ok ? tap : 0;
    int dy = tp / 3 - 1, dx = tp - (tp / 3) * 3 - 1;
    int pix = w * 16 + n;
    int rl = pix >> 3, c = pix & 7;
    int cc = c + dx;
    int ccc = cc < 0 ? 0 : (cc > 7 ? 7 : cc);
    unsigned short v = tile[ic][rl + 1 + dy][ccc];
    bmat[w][s][n][kk] = (ok && cc == ccc) ? v : (unsigned short)0;
  }
  __syncthreads();

  // ---- phase 3: WMMA main loop, 4x ds_load_b128 + wmma per k-step ----
  int w = threadIdx.x >> 5;
  int lane = threadIdx.x & 31;
  int n = lane & 15, g = lane >> 4;
  v8f acc = {};
#pragma unroll
  for (int s = 0; s < 3; ++s) {
    // half j<8 -> K = g*8 + j ; half j>=8 -> K = 16 + g*8 + (j-8): two 16B runs
    Pack32 ap, bp;
    ap.a = *(const uint4*)&wA[s][n][g * 8];
    ap.b = *(const uint4*)&wA[s][n][g * 8 + 16];
    bp.a = *(const uint4*)&bmat[w][s][n][g * 8];
    bp.b = *(const uint4*)&bmat[w][s][n][g * 8 + 16];
    v16bf Af = __builtin_bit_cast(v16bf, ap);
    v16bf Bf = __builtin_bit_cast(v16bf, bp);
    acc = __builtin_amdgcn_wmma_f32_16x16x32_bf16(
        false, Af, false, Bf, (short)0, acc, false, false);
  }
  // D: lane holds column N = lane&15 (pixel), VGPR v -> channel v + 8*g
  int p = r0 * 8 + w * 16 + n;
  unsigned short* out = mem2 + (size_t)b * 16 * NTOT;
#pragma unroll
  for (int v = 0; v < 8; ++v) {
    int mo = v + 8 * g;
    float val = acc[v] + b2[mo];
    val = val > 0.f ? val : 0.f;
    out[(size_t)mo * NTOT + p] = f2bf(val);
  }
}

// ---------------- adaptive avg pool (4,4) ----------------
__global__ __launch_bounds__(256) void pool_kernel(
    const unsigned short* __restrict__ mem2, unsigned short* __restrict__ flatb) {
  int out = blockIdx.x;                        // b*256 + c*16 + ph*4 + pw
  int b = out >> 8, rem = out & 255;
  int c = rem >> 4, ph = (rem >> 2) & 3, pw = rem & 3;
  const unsigned short* src = mem2 + ((size_t)b * 16 + c) * NTOT;
  float sum = 0.f;
  for (int i = threadIdx.x; i < NROWS / 4; i += 256) {
    size_t off = (size_t)(ph * (NROWS / 4) + i) * 8 + pw * 2;
    sum += bf2f(src[off]) + bf2f(src[off + 1]);
  }
  __shared__ float red[256];
  red[threadIdx.x] = sum;
  __syncthreads();
  for (int st = 128; st > 0; st >>= 1) {
    if (threadIdx.x < st) red[threadIdx.x] += red[threadIdx.x + st];
    __syncthreads();
  }
  if (threadIdx.x == 0)
    flatb[b * 256 + (c * 16 + ph * 4 + pw)] = f2bf(red[0] / (float)((NROWS / 4) * 2));
}

__global__ void zero_flat_tail_kernel(unsigned short* __restrict__ flatb) {
  flatb[(8 + blockIdx.x) * 256 + threadIdx.x] = 0;   // zero M rows 8..15
}

// ---------------- projection GEMM [16x256] x [256 x 65536] via WMMA ----------------
__global__ __launch_bounds__(256) void proj_wmma_kernel(
    const unsigned short* __restrict__ flatb,
    const float* __restrict__ pw, const float* __restrict__ pb,
    float* __restrict__ logits) {
  int wave = blockIdx.x * (blockDim.x >> 5) + (threadIdx.x >> 5);  // 4096 waves
  int lane = threadIdx.x & 31;
  int n = lane & 15, g = lane >> 4;
  int o0 = wave * 16;
  const float* rowp = pw + (size_t)(o0 + n) * 256;     // B column N = lane&15
  const unsigned short* frow = flatb + n * 256;        // A row    M = lane&15
  v8f acc = {};
#pragma unroll
  for (int s = 0; s < 8; ++s) {
    int off = 32 * s + g * 8;
    Pack32 ap;
    ap.a = *(const uint4*)(frow + off);                // halves K=off..off+7
    ap.b = *(const uint4*)(frow + off + 16);           // halves K=off+16..off+23
    v16bf Af = __builtin_bit_cast(v16bf, ap);
    float4 f0 = *(const float4*)(rowp + off);
    float4 f1 = *(const float4*)(rowp + off + 4);
    float4 f2 = *(const float4*)(rowp + off + 16);
    float4 f3 = *(const float4*)(rowp + off + 20);
    v16bf Bf;
    Bf[0]  = us2bf(f2bf(f0.x)); Bf[1]  = us2bf(f2bf(f0.y));
    Bf[2]  = us2bf(f2bf(f0.z)); Bf[3]  = us2bf(f2bf(f0.w));
    Bf[4]  = us2bf(f2bf(f1.x)); Bf[5]  = us2bf(f2bf(f1.y));
    Bf[6]  = us2bf(f2bf(f1.z)); Bf[7]  = us2bf(f2bf(f1.w));
    Bf[8]  = us2bf(f2bf(f2.x)); Bf[9]  = us2bf(f2bf(f2.y));
    Bf[10] = us2bf(f2bf(f2.z)); Bf[11] = us2bf(f2bf(f2.w));
    Bf[12] = us2bf(f2bf(f3.x)); Bf[13] = us2bf(f2bf(f3.y));
    Bf[14] = us2bf(f2bf(f3.z)); Bf[15] = us2bf(f2bf(f3.w));
    acc = __builtin_amdgcn_wmma_f32_16x16x32_bf16(
        false, Af, false, Bf, (short)0, acc, false, false);
  }
  if (g == 0) {                                        // rows 0..7 = real batches
#pragma unroll
    for (int v = 0; v < 8; ++v)
      logits[(size_t)v * NOPSc + o0 + n] = acc[v] + pb[o0 + n];
  }
}

// ---------------- staged penalties ----------------
__device__ __forceinline__ float stagef(float h) {
  return h <= 2.f ? 1.f : h <= 4.f ? 1.5f : h <= 8.f ? 2.f : h <= 16.f ? 3.f : 5.f;
}
__device__ __forceinline__ float pen(float d) {
  if (d > 0.f) return d * stagef(d);
  if (d < 0.f) { float h = -d; return h * h * stagef(h); }
  return 0.f;
}

__global__ __launch_bounds__(256) void penalty_kernel(
    const unsigned int* __restrict__ addr, const unsigned int* __restrict__ perm,
    float* __restrict__ partials) {
  int b = blockIdx.y;
  int r = blockIdx.x * 256 + threadIdx.x;
  const unsigned int* ab = addr + (size_t)b * NTOT;
  const unsigned int* pm = perm + (size_t)b * NOPSc;
  unsigned int t = pm[r];
  float s0 = (float)ab[t];
  float s1 = (float)ab[NOPSc + t];
  float s2 = (float)ab[2 * NOPSc + t];
  float intra = pen(s1 - s0) + pen(s2 - s1);
  float inter = 0.f;
  if (r + 1 < NOPSc) {
    unsigned int tn = pm[r + 1];
    inter = pen((float)ab[tn] - s2);
  }
  __shared__ float ri[256], rj[256];
  ri[threadIdx.x] = inter;
  rj[threadIdx.x] = intra;
  __syncthreads();
  for (int st = 128; st > 0; st >>= 1) {
    if (threadIdx.x < st) {
      ri[threadIdx.x] += ri[threadIdx.x + st];
      rj[threadIdx.x] += rj[threadIdx.x + st];
    }
    __syncthreads();
  }
  if (threadIdx.x == 0) {
    partials[((size_t)b * 256 + blockIdx.x) * 2 + 0] = ri[0];
    partials[((size_t)b * 256 + blockIdx.x) * 2 + 1] = rj[0];
  }
}

__global__ __launch_bounds__(256) void finalize_kernel(
    const float* __restrict__ partials, float* __restrict__ out) {
  int b = blockIdx.x;
  __shared__ float ri[256], rj[256];
  ri[threadIdx.x] = partials[((size_t)b * 256 + threadIdx.x) * 2 + 0];
  rj[threadIdx.x] = partials[((size_t)b * 256 + threadIdx.x) * 2 + 1];
  __syncthreads();
  for (int st = 128; st > 0; st >>= 1) {
    if (threadIdx.x < st) {
      ri[threadIdx.x] += ri[threadIdx.x + st];
      rj[threadIdx.x] += rj[threadIdx.x + st];
    }
    __syncthreads();
  }
  if (threadIdx.x == 0) {
    out[b]     = ri[0];
    out[8 + b] = rj[0];
  }
}

// ---------------- host orchestration ----------------
static inline unsigned char* carve(unsigned char*& p, size_t bytes) {
  unsigned char* q = p;
  p += (bytes + 255) & ~(size_t)255;
  return q;
}

static inline void run_bitonic(unsigned long long* keys, int len, int total,
                               hipStream_t stream) {
  bitonic_sort_local_kernel<<<total / LOCELEMS, 256, 0, stream>>>(keys, len);
  for (unsigned int k = LOCELEMS * 2; k <= (unsigned)len; k <<= 1) {
    for (unsigned int j = k >> 1; j >= LOCELEMS; j >>= 1)
      bitonic_step_kernel<<<total / 256, 256, 0, stream>>>(keys, len, j, k);
    bitonic_merge_local_kernel<<<total / LOCELEMS, 256, 0, stream>>>(keys, len, k);
  }
}

extern "C" void kernel_launch(void* const* d_in, const int* in_sizes, int n_in,
                              void* d_out, int out_size, void* d_ws, size_t ws_size,
                              hipStream_t stream) {
  const float* mem_logits = (const float*)d_in[0];
  const float* conv1_w    = (const float*)d_in[1];
  const float* conv1_b    = (const float*)d_in[2];
  const float* conv2_w    = (const float*)d_in[3];
  const float* conv2_b    = (const float*)d_in[4];
  const float* proj_w     = (const float*)d_in[5];
  const float* proj_b     = (const float*)d_in[6];
  float* out = (float*)d_out;

  unsigned char* p = (unsigned char*)d_ws;
  unsigned long long* keys1 = (unsigned long long*)carve(p, (size_t)Bc * LEN1 * 8);
  unsigned long long* keys2 = (unsigned long long*)carve(p, (size_t)Bc * LEN2 * 8);
  unsigned int* addr  = (unsigned int*)carve(p, (size_t)Bc * NTOT * 4);
  unsigned int* perm  = (unsigned int*)carve(p, (size_t)Bc * NOPSc * 4);
  unsigned short* mem  = (unsigned short*)carve(p, (size_t)Bc * 8 * NTOT * 2);
  unsigned short* mem2 = (unsigned short*)carve(p, (size_t)Bc * 16 * NTOT * 2);
  float* op_logits = (float*)carve(p, (size_t)Bc * NOPSc * 4);
  unsigned short* flatb = (unsigned short*)carve(p, 16 * 256 * 2);
  float* partials = (float*)carve(p, (size_t)Bc * 256 * 2 * 4);
  (void)ws_size; (void)in_sizes; (void)n_in; (void)out_size;

  // 1) argsort(mem_logits): LDS bitonic + short global ladder
  init_keys1_kernel<<<(Bc * LEN1) / 256, 256, 0, stream>>>(mem_logits, keys1);
  run_bitonic(keys1, LEN1, Bc * LEN1, stream);
  extract_lo_kernel<<<((size_t)Bc * NTOT) / 256, 256, 0, stream>>>(keys1, addr, LEN1, NTOT);

  // 2) conv1 + permutation scatter into memory grid (bf16)
  conv1_scatter_kernel<<<((size_t)Bc * Kc * HWc) / 256, 256, 0, stream>>>(
      addr, conv1_w, conv1_b, mem);

  // 3) conv2: async-LDS-staged implicit GEMM, v_wmma_f32_16x16x32_bf16
  conv2_wmma_kernel<<<Bc * (NROWS / 16), 256, 0, stream>>>(mem, conv2_w, conv2_b, mem2);

  // 4) adaptive avg-pool 4x4 -> bf16 activation rows
  zero_flat_tail_kernel<<<8, 256, 0, stream>>>(flatb);
  pool_kernel<<<Bc * 16 * 16, 256, 0, stream>>>(mem2, flatb);

  // 5) projection GEMM via WMMA -> op_logits
  proj_wmma_kernel<<<(NOPSc / 16) / 8, 256, 0, stream>>>(flatb, proj_w, proj_b, op_logits);

  // 6) argsort(op_logits)
  init_keys2_kernel<<<(Bc * LEN2) / 256, 256, 0, stream>>>(op_logits, keys2);
  run_bitonic(keys2, LEN2, Bc * LEN2, stream);
  extract_lo_kernel<<<((size_t)Bc * NOPSc) / 256, 256, 0, stream>>>(keys2, perm, LEN2, NOPSc);

  // 7) staged penalties + deterministic reduction -> d_out[16]
  penalty_kernel<<<dim3(NOPSc / 256, Bc), 256, 0, stream>>>(addr, perm, partials);
  finalize_kernel<<<Bc, 256, 0, stream>>>(partials, out);
}
